// MultiHeadAttention_61332132987729
// MI455X (gfx1250) — compile-verified, matrix operand order checked by reading, not run on hardware
//
#include <hip/hip_runtime.h>
#include <hip/hip_bf16.h>

#define D_MODELc 1024
#define D_Kc 64
#define D_Vc 64
#define N_HEADSc 16
#define BBc 2
#define SSc 2048

typedef __attribute__((ext_vector_type(16))) __bf16 bf16x16;
typedef __attribute__((ext_vector_type(8)))  __bf16 bf16x8;
typedef __attribute__((ext_vector_type(8)))  float  f32x8;
typedef __attribute__((ext_vector_type(4)))  unsigned int u32x4;
typedef __attribute__((ext_vector_type(8)))  int i32x8;
typedef __attribute__((ext_vector_type(4)))  int i32x4;

// ---- helpers -------------------------------------------------------------

__device__ __forceinline__ unsigned short f2bf(float f) {
  unsigned int u = __float_as_uint(f);
  u += 0x7FFFu + ((u >> 16) & 1u);   // round-to-nearest-even
  return (unsigned short)(u >> 16);
}

__device__ __forceinline__ bf16x8 ld8(const unsigned short* p) {
  return *reinterpret_cast<const bf16x8*>(p);
}

// Load one bf16 WMMA tile (A: 16x32 MxK, or B: 32x16 KxN by columns).
// ISA layout: lane<16 -> row/col = lane, elems 0..7 = K[kh*8 .. +7],
// elems 8..15 = K[16+kh*8 .. +7], where kh = lane>>4.
__device__ __forceinline__ bf16x16 load_tile(const unsigned short* src, int ld, int kbase) {
  int lane = threadIdx.x & 31;
  int m  = lane & 15;
  int kh = lane >> 4;
  const unsigned short* p = src + (size_t)m * ld + kbase + kh * 8;
  bf16x8 lo = ld8(p);
  bf16x8 hi = ld8(p + 16);
  bf16x16 r;
  #pragma unroll
  for (int i = 0; i < 8; ++i) { r[i] = lo[i]; r[i + 8] = hi[i]; }
  return r;
}

__device__ __forceinline__ f32x8 wmma_bf16(bf16x16 a, bf16x16 b, f32x8 c) {
  return __builtin_amdgcn_wmma_f32_16x16x32_bf16(false, a, false, b, (short)0, c, false, false);
}

__device__ __forceinline__ float rowmax16(float v) {
  #pragma unroll
  for (int m = 8; m >= 1; m >>= 1) v = fmaxf(v, __shfl_xor(v, m, 32));
  return v;
}
__device__ __forceinline__ float rowsum16(float v) {
  #pragma unroll
  for (int m = 8; m >= 1; m >>= 1) v += __shfl_xor(v, m, 32);
  return v;
}

// ---- Tensor Data Mover: 2D tile (16-bit elements) global -> LDS -----------
// D# group0: count=1 | lds_addr | global_addr | type=2
// D# group1: data_size=2B; tensor_dim0/1, tile_dim0/1, tensor_dim0_stride
// Dims/strides are in elements; lds_addr/global addr in bytes.
__device__ __forceinline__ void tdm_load_2d(const void* gptr, unsigned lds_addr,
                                            unsigned tile_d0, unsigned tile_d1,
                                            unsigned tensor_d0, unsigned tensor_d1,
                                            unsigned long long stride0) {
  unsigned long long ga = (unsigned long long)(size_t)gptr;
  u32x4 g0;
  g0[0] = 1u;                                               // count=1, user desc
  g0[1] = lds_addr;                                         // bits 63:32
  g0[2] = (unsigned)(ga & 0xFFFFFFFFu);                     // global addr lo
  g0[3] = (unsigned)((ga >> 32) & 0x01FFFFFFu) | 0x80000000u; // addr[56:32] | type=2
  i32x8 g1;
  g1[0] = (int)(1u << 16);                                  // data_size = 2 bytes
  g1[1] = (int)((tensor_d0 & 0xFFFFu) << 16);               // tensor_dim0[15:0] at bit48
  g1[2] = (int)(((tensor_d0 >> 16) & 0xFFFFu) | ((tensor_d1 & 0xFFFFu) << 16));
  g1[3] = (int)(((tensor_d1 >> 16) & 0xFFFFu) | ((tile_d0 & 0xFFFFu) << 16));
  g1[4] = (int)(tile_d1 & 0xFFFFu);                         // tile_dim1 (tile_dim2=0)
  g1[5] = (int)(unsigned)(stride0 & 0xFFFFFFFFull);         // dim0 stride lo
  g1[6] = (int)(unsigned)((stride0 >> 32) & 0xFFFFull);     // dim0 stride hi (dim1 stride=0)
  g1[7] = 0;
  i32x4 z4 = {0, 0, 0, 0};
  i32x8 z8 = {0, 0, 0, 0, 0, 0, 0, 0};
  // clang-23 / therock-10.0 form: (g0, g1, g2, g3, g4, cpol)
  __builtin_amdgcn_tensor_load_to_lds(g0, g1, z4, z4, z8, 0);
}

__device__ __forceinline__ unsigned lds_offset_of(const void* p) {
  return (unsigned)(size_t)p;   // flat shared addr[31:0] == LDS offset
}

// ---- conversion kernels ---------------------------------------------------

__global__ void cvt_kernel(const float* __restrict__ src, unsigned short* __restrict__ dst,
                           long long n) {
  long long i = (long long)blockIdx.x * blockDim.x + threadIdx.x;
  long long stride = (long long)gridDim.x * blockDim.x;
  for (; i < n; i += stride) dst[i] = f2bf(src[i]);
}

// src [Hn, Dn, Kn] f32 -> dst [Hn, Kn, Dn] bf16
__global__ void cvt_transpose_kernel(const float* __restrict__ src, unsigned short* __restrict__ dst,
                                     int Hn, int Dn, int Kn) {
  long long n = (long long)Hn * Dn * Kn;
  long long i = (long long)blockIdx.x * blockDim.x + threadIdx.x;
  long long stride = (long long)gridDim.x * blockDim.x;
  for (; i < n; i += stride) {
    int k = (int)(i % Kn);
    long long t = i / Kn;
    int d = (int)(t % Dn);
    int h = (int)(t / Dn);
    dst[((size_t)h * Kn + k) * Dn + d] = f2bf(src[i]);
  }
}

// ---- per-head projection: P[b,h,s,k] = X[b,s,:] . W[h,:,k] + bias[h,k] ----

__global__ __launch_bounds__(128) void proj_kernel(const unsigned short* __restrict__ X,
                                                   const unsigned short* __restrict__ WT,
                                                   const float* __restrict__ bias,
                                                   unsigned short* __restrict__ P,
                                                   int transpose_out) {
  int wave = threadIdx.x >> 5, lane = threadIdx.x & 31;
  int colA = lane & 15, hi = lane >> 4;
  int bh = blockIdx.y;
  int b = bh >> 4, h = bh & 15;
  int row0 = (blockIdx.x * 4 + wave) * 16;

  const unsigned short* Xr = X + ((size_t)b * SSc + row0) * D_MODELc;
  const unsigned short* Wh = WT + (size_t)h * D_Kc * D_MODELc;

  f32x8 zero = {};
  f32x8 acc[4] = {zero, zero, zero, zero};

  for (int d = 0; d < D_MODELc; d += 32) {
    bf16x16 a = load_tile(Xr, D_MODELc, d);
    #pragma unroll
    for (int nt = 0; nt < 4; ++nt)
      acc[nt] = wmma_bf16(a, load_tile(Wh + (size_t)(nt * 16) * D_MODELc, D_MODELc, d), acc[nt]);
  }

  #pragma unroll
  for (int nt = 0; nt < 4; ++nt) {
    #pragma unroll
    for (int r = 0; r < 8; ++r) {
      int row = row0 + r + 8 * hi;
      int col = nt * 16 + colA;
      float v = acc[nt][r] + bias[h * D_Kc + col];
      size_t idx = transpose_out ? ((size_t)bh * D_Kc + col) * SSc + row
                                 : ((size_t)bh * SSc + row) * D_Kc + col;
      P[idx] = f2bf(v);
    }
  }
}

// ---- flash attention with TDM-staged K/V ----------------------------------
// Block = 4 waves sharing one (b,h): K/V 64-key chunks are DMA'd into LDS
// once per block (double-buffered, TENSORcnt-synced), all score/ctx WMMA
// B-tiles then come from LDS. Final scale 1/(l*128) matches softmax/d_k/2.

__global__ __launch_bounds__(128) void attn_kernel(const unsigned short* __restrict__ QW,   // [B,H,S,64]
                                                   const unsigned short* __restrict__ KW,   // [B,H,S,64]
                                                   const unsigned short* __restrict__ VWt,  // [B,H,64,S]
                                                   unsigned short* __restrict__ CTX) {      // [B,S,H*64]
  __shared__ __align__(16) unsigned short lds_p[4][16 * 32];
  __shared__ __align__(16) unsigned short ldsK[2][64 * 64];   // [t(64)][k(64)] chunk
  __shared__ __align__(16) unsigned short ldsV[2][64 * 64];   // [v(64)][t(64)] chunk

  int wave = threadIdx.x >> 5, lane = threadIdx.x & 31;
  int colA = lane & 15, hi = lane >> 4;
  int bh = blockIdx.y;
  int b = bh >> 4, h = bh & 15;
  int row0 = (blockIdx.x * 4 + wave) * 16;
  bool issuer = (wave == 0);

  const unsigned short* QWb = QW + ((size_t)bh * SSc + row0) * D_Kc;
  const unsigned short* KWb = KW + (size_t)bh * SSc * D_Kc;
  const unsigned short* VWb = VWt + (size_t)bh * D_Vc * SSc;

  bf16x16 a0 = load_tile(QWb, D_Kc, 0);
  bf16x16 a1 = load_tile(QWb, D_Kc, 32);

  f32x8 zero = {};
  f32x8 o0 = zero, o1 = zero, o2 = zero, o3 = zero;
  float m[8], l[8];
  #pragma unroll
  for (int r = 0; r < 8; ++r) { m[r] = -3.0e38f; l[r] = 0.0f; }

  unsigned short* lp = &lds_p[wave][0];
  const int NC = SSc / 64;

  // prologue: DMA chunk 0
  if (issuer) {
    tdm_load_2d(KWb, lds_offset_of(&ldsK[0][0]), 4096, 1, 1u << 20, 1, 4096ull);
    tdm_load_2d(VWb, lds_offset_of(&ldsV[0][0]), 64, 64, SSc, 64, (unsigned long long)SSc);
  }

  for (int c = 0; c < NC; ++c) {
    if (issuer) __builtin_amdgcn_s_wait_tensorcnt(0);
    __syncthreads();   // chunk c resident; all waves done with buffer (c+1)&1
    if (issuer && (c + 1 < NC)) {
      int tn = (c + 1) * 64;
      tdm_load_2d(KWb + (size_t)tn * D_Kc, lds_offset_of(&ldsK[(c + 1) & 1][0]),
                  4096, 1, 1u << 20, 1, 4096ull);
      tdm_load_2d(VWb + tn, lds_offset_of(&ldsV[(c + 1) & 1][0]),
                  64, 64, SSc, 64, (unsigned long long)SSc);
    }

    const unsigned short* Kc = &ldsK[c & 1][0];
    const unsigned short* Vc = &ldsV[c & 1][0];

    #pragma unroll
    for (int sub = 0; sub < 64; sub += 32) {
      f32x8 s0 = zero, s1 = zero;
      s0 = wmma_bf16(a0, load_tile(Kc + (size_t)(sub) * D_Kc, D_Kc, 0), s0);
      s0 = wmma_bf16(a1, load_tile(Kc + (size_t)(sub) * D_Kc, D_Kc, 32), s0);
      s1 = wmma_bf16(a0, load_tile(Kc + (size_t)(sub + 16) * D_Kc, D_Kc, 0), s1);
      s1 = wmma_bf16(a1, load_tile(Kc + (size_t)(sub + 16) * D_Kc, D_Kc, 32), s1);

      float al[8];
      #pragma unroll
      for (int r = 0; r < 8; ++r) {
        float v0 = s0[r], v1 = s1[r];
        float mx = rowmax16(fmaxf(v0, v1));
        float mn = fmaxf(m[r], mx);
        float a  = __expf(m[r] - mn);
        float p0 = __expf(v0 - mn);
        float p1 = __expf(v1 - mn);
        float sum = rowsum16(p0 + p1);
        l[r] = l[r] * a + sum;
        m[r] = mn;
        al[r] = a;
        int row = r + 8 * hi;
        lp[row * 32 + colA]      = f2bf(p0);
        lp[row * 32 + 16 + colA] = f2bf(p1);
      }
      #pragma unroll
      for (int r = 0; r < 8; ++r) { o0[r] *= al[r]; o1[r] *= al[r]; o2[r] *= al[r]; o3[r] *= al[r]; }

      asm volatile("" ::: "memory");
      bf16x16 ap = load_tile(lp, 32, 0);     // P: C-layout -> A-layout via per-wave LDS
      asm volatile("" ::: "memory");

      o0 = wmma_bf16(ap, load_tile(Vc + (size_t)(0)  * 64 + sub, 64, 0), o0);
      o1 = wmma_bf16(ap, load_tile(Vc + (size_t)(16) * 64 + sub, 64, 0), o1);
      o2 = wmma_bf16(ap, load_tile(Vc + (size_t)(32) * 64 + sub, 64, 0), o2);
      o3 = wmma_bf16(ap, load_tile(Vc + (size_t)(48) * 64 + sub, 64, 0), o3);
    }
  }

  #pragma unroll
  for (int r = 0; r < 8; ++r) {
    float inv = 1.0f / (l[r] * 128.0f);
    int row = row0 + r + 8 * hi;
    unsigned short* dst = CTX + (((size_t)b * SSc + row) * N_HEADSc + h) * D_Vc;
    dst[ 0 + colA] = f2bf(o0[r] * inv);
    dst[16 + colA] = f2bf(o1[r] * inv);
    dst[32 + colA] = f2bf(o2[r] * inv);
    dst[48 + colA] = f2bf(o3[r] * inv);
  }
}

// ---- output projection: out = ctx[4096,1024] . Wo[1024,1024] + bo ---------

__global__ __launch_bounds__(128) void outproj_kernel(const unsigned short* __restrict__ CTX,
                                                      const unsigned short* __restrict__ WoT,
                                                      const float* __restrict__ bo,
                                                      float* __restrict__ out) {
  int wave = threadIdx.x >> 5, lane = threadIdx.x & 31;
  int colA = lane & 15, hi = lane >> 4;
  int row0 = (blockIdx.x * 4 + wave) * 16;
  int cb = blockIdx.y * 64;

  const unsigned short* Ar = CTX + (size_t)row0 * D_MODELc;
  f32x8 zero = {};
  f32x8 acc[4] = {zero, zero, zero, zero};

  for (int d = 0; d < D_MODELc; d += 32) {
    bf16x16 a = load_tile(Ar, D_MODELc, d);
    #pragma unroll
    for (int nt = 0; nt < 4; ++nt)
      acc[nt] = wmma_bf16(a, load_tile(WoT + (size_t)(cb + nt * 16) * D_MODELc, D_MODELc, d), acc[nt]);
  }

  #pragma unroll
  for (int nt = 0; nt < 4; ++nt) {
    #pragma unroll
    for (int r = 0; r < 8; ++r) {
      int row = row0 + r + 8 * hi;
      int col = cb + nt * 16 + colA;
      out[(size_t)row * D_MODELc + col] = acc[nt][r] + bo[col];
    }
  }
}

// ---- launch ----------------------------------------------------------------

extern "C" void kernel_launch(void* const* d_in, const int* in_sizes, int n_in,
                              void* d_out, int out_size, void* d_ws, size_t ws_size,
                              hipStream_t stream) {
  (void)in_sizes; (void)n_in; (void)out_size; (void)ws_size;

  const float* Qf = (const float*)d_in[0];
  const float* Kf = (const float*)d_in[1];
  const float* Vf = (const float*)d_in[2];
  const float* Wq = (const float*)d_in[3];
  const float* bq = (const float*)d_in[4];
  const float* Wk = (const float*)d_in[5];
  const float* bk = (const float*)d_in[6];
  const float* Wv = (const float*)d_in[7];
  const float* bv = (const float*)d_in[8];
  const float* Wo = (const float*)d_in[9];
  const float* bo = (const float*)d_in[10];

  const size_t RD  = (size_t)BBc * SSc * D_MODELc;
  const size_t WSZ = (size_t)N_HEADSc * D_MODELc * D_Kc;
  const size_t PSZ = (size_t)BBc * N_HEADSc * SSc * D_Kc;

  unsigned short* ws  = (unsigned short*)d_ws;
  unsigned short* Xq  = ws;
  unsigned short* Xk  = Xq + RD;
  unsigned short* Xv  = Xk + RD;
  unsigned short* WqT = Xv + RD;
  unsigned short* WkT = WqT + WSZ;
  unsigned short* WvT = WkT + WSZ;
  unsigned short* WoT = WvT + WSZ;
  unsigned short* QWp = WoT + (size_t)D_MODELc * D_MODELc;
  unsigned short* KWp = QWp + PSZ;
  unsigned short* VWt = KWp + PSZ;
  unsigned short* CTX = VWt + PSZ;

  cvt_kernel<<<2048, 256, 0, stream>>>(Qf, Xq, (long long)RD);
  cvt_kernel<<<2048, 256, 0, stream>>>(Kf, Xk, (long long)RD);
  cvt_kernel<<<2048, 256, 0, stream>>>(Vf, Xv, (long long)RD);
  cvt_transpose_kernel<<<2048, 256, 0, stream>>>(Wq, WqT, N_HEADSc, D_MODELc, D_Kc);
  cvt_transpose_kernel<<<2048, 256, 0, stream>>>(Wk, WkT, N_HEADSc, D_MODELc, D_Kc);
  cvt_transpose_kernel<<<2048, 256, 0, stream>>>(Wv, WvT, N_HEADSc, D_MODELc, D_Kc);
  cvt_transpose_kernel<<<2048, 256, 0, stream>>>(Wo, WoT, 1, D_MODELc, D_MODELc);

  dim3 pg(SSc / 64, BBc * N_HEADSc);
  proj_kernel<<<pg, 128, 0, stream>>>(Xq, WqT, bq, QWp, 0);
  proj_kernel<<<pg, 128, 0, stream>>>(Xk, WkT, bk, KWp, 0);
  proj_kernel<<<pg, 128, 0, stream>>>(Xv, WvT, bv, VWt, 1);

  attn_kernel<<<dim3(SSc / 64, BBc * N_HEADSc), 128, 0, stream>>>(QWp, KWp, VWt, CTX);

  outproj_kernel<<<dim3((BBc * SSc) / 64, D_MODELc / 64), 128, 0, stream>>>(CTX, WoT, bo,
                                                                            (float*)d_out);
}
